// Net_76665166234219
// MI455X (gfx1250) — matrix operand order repaired and checked by reading, not compiled, not run on hardware
//
#include <hip/hip_runtime.h>
#include <hip/hip_bf16.h>
#include <math.h>

#define ALPHA 0.4f
#define THETA 0.9f
#define H 64

typedef __attribute__((ext_vector_type(16))) __bf16 v16bf;
typedef __attribute__((ext_vector_type(8)))  float  v8f;

// ---------------- helpers ----------------

static __device__ __forceinline__ __bf16 f2bf(float f) {
    return (__bf16)f;   // RNE, lowers to v_cvt_pk_bf16_f32 pairs
}

static __device__ __forceinline__ float4 ld4(const float* p) {
    return *(const float4*)p;   // global_load_b128
}

static __device__ __forceinline__ v8f wmma_bf16(v16bf a, v16bf b, v8f c) {
    // D = A(16x32 bf16) x B(32x16 bf16) + C(16x16 f32)
    return __builtin_amdgcn_wmma_f32_16x16x32_bf16(
        /*neg_a=*/false, a, /*neg_b=*/false, b,
        /*c_mod=*/(short)0, c, /*reuse_a=*/false, /*reuse_b=*/false);
}

// A fragment 16x32 (MxK) from row-major f32, K%32==0, rows 16-float aligned.
// Contiguous-k layout (same as k_pack_b): lane(0..15) row M=lane&15; elems 0..15 = K{half*16..+15}
static __device__ __forceinline__ v16bf
load_a(const float* __restrict__ X, int lda, int m0, int k0,
       int n_rows, int lane, float scale) {
    int r = m0 + (lane & 15);
    if (r >= n_rows) r = n_rows - 1;          // safety clamp (no-op when N%32==0)
    const float* p = X + (long)r * lda + k0 + ((lane >> 4) * 16);
    float4 q0 = ld4(p), q1 = ld4(p + 4), q2 = ld4(p + 8), q3 = ld4(p + 12);
    v16bf a;
    a[0]  = f2bf(scale * q0.x); a[1]  = f2bf(scale * q0.y);
    a[2]  = f2bf(scale * q0.z); a[3]  = f2bf(scale * q0.w);
    a[4]  = f2bf(scale * q1.x); a[5]  = f2bf(scale * q1.y);
    a[6]  = f2bf(scale * q1.z); a[7]  = f2bf(scale * q1.w);
    a[8]  = f2bf(scale * q2.x); a[9]  = f2bf(scale * q2.y);
    a[10] = f2bf(scale * q2.z); a[11] = f2bf(scale * q2.w);
    a[12] = f2bf(scale * q3.x); a[13] = f2bf(scale * q3.y);
    a[14] = f2bf(scale * q3.z); a[15] = f2bf(scale * q3.w);
    return a;
}

// Packed-B fragment load: 16 bf16 per lane, fragment-major layout (2x b128 per lane)
static __device__ __forceinline__ v16bf
load_bp(const __bf16* __restrict__ P, int frag, int lane) {
    return *(const v16bf*)(P + (((long)frag * 32 + lane) << 4));
}

// ---------------- kernels ----------------

// Pack (bscale * W + dadd * I) into WMMA B-fragment order (bf16).
// Logical B is [K,64]; fragment f = kb*4 + t holds B[kb*32..+31][t*16..+15].
// Element (f, lane, i): k = kb*32 + (lane>>4)*16 + i, n = t*16 + (lane&15).
// trans=1: source W is [64,K] row-major (NT gemm, B[k][n]=W[n][k]);
// trans=0: source W is [K,64] row-major (NN gemm).
__global__ void k_pack_b(const float* __restrict__ W, __bf16* __restrict__ P,
                         int K, int trans, float bscale, float dadd) {
    int g = blockIdx.x * blockDim.x + threadIdx.x;
    if (g >= K * H) return;
    int i    = g & 15;
    int lane = (g >> 4) & 31;
    int f    = g >> 9;
    int kb = f >> 2, t = f & 3;
    int k = kb * 32 + ((lane >> 4) << 4) + i;
    int n = t * 16 + (lane & 15);
    float v = trans ? W[(long)n * K + k] : W[(long)k * H + n];
    v = bscale * v + ((k == n) ? dadd : 0.0f);
    P[g] = (__bf16)v;
}

// dst[r, 0..Kout) = (k < Kin) ? src[r, k] : 0   (row zero-padding)
__global__ void k_pad_rows(const float* __restrict__ src, float* __restrict__ dst,
                           long n_rows, int Kin, int Kout) {
    long g = (long)blockIdx.x * blockDim.x + threadIdx.x;
    long total = n_rows * Kout;
    if (g >= total) return;
    int k = (int)(g % Kout);
    long r = g / Kout;
    dst[g] = (k < Kin) ? src[r * Kin + k] : 0.0f;
}

__global__ void k_degree(const int* __restrict__ dst, float* __restrict__ deg, long E) {
    long e = (long)blockIdx.x * blockDim.x + threadIdx.x;
    if (e < E) atomicAdd(&deg[dst[e]], 1.0f);
}

__global__ void k_dinv(float* __restrict__ deg, int n) {
    int i = blockIdx.x * blockDim.x + threadIdx.x;
    if (i < n) {
        float d = deg[i];
        deg[i] = (d > 0.0f) ? rsqrtf(d) : 0.0f;
    }
}

__global__ void k_norm(const int* __restrict__ src, const int* __restrict__ dst,
                       const float* __restrict__ dinv, float* __restrict__ nrm, long E) {
    long e = (long)blockIdx.x * blockDim.x + threadIdx.x;
    if (e < E) nrm[e] = dinv[src[e]] * dinv[dst[e]];
}

// agg[dst] += norm * h[src] ; 16 lanes per edge, float4 per lane (coalesced 256B gather)
__global__ void k_scatter(const int* __restrict__ src, const int* __restrict__ dst,
                          const float* __restrict__ nrm, const float* __restrict__ hin,
                          float* __restrict__ agg, long E) {
    long g = (long)blockIdx.x * blockDim.x + threadIdx.x;
    long e = g >> 4;
    if (e >= E) return;
    int t = (int)(g & 15);
    int s = src[e], d = dst[e];
    float w = nrm[e];
    float4 v = ((const float4*)(hin + (long)s * H))[t];
    float* ad = agg + (long)d * H + t * 4;
    atomicAdd(ad + 0, w * v.x);
    atomicAdd(ad + 1, w * v.y);
    atomicAdd(ad + 2, w * v.z);
    atomicAdd(ad + 3, w * v.w);
}

// Hout = relu(X[n_rows,K] @ B + bias) with B pre-packed bf16; K%32==0.
// One wave computes a 32x64 output tile (two 16-row subtiles sharing B fragments).
__global__ void __launch_bounds__(256)
k_gemm_relu(const float* __restrict__ X, const __bf16* __restrict__ Pb,
            const float* __restrict__ bias, float* __restrict__ Hout,
            int n_rows, int K) {
    int wave = blockIdx.x * (blockDim.x >> 5) + (threadIdx.x >> 5);
    int lane = threadIdx.x & 31;
    int m0 = wave * 32;
    if (m0 >= n_rows) return;              // wave-uniform: EXEC stays all-ones
    v8f acc[2][4] = {};
    int nkb = K >> 5;
    for (int kb = 0; kb < nkb; ++kb) {
        v16bf a0 = load_a(X, K, m0,      kb * 32, n_rows, lane, 1.0f);
        v16bf a1 = load_a(X, K, m0 + 16, kb * 32, n_rows, lane, 1.0f);
#pragma unroll
        for (int t = 0; t < 4; ++t) {
            v16bf b = load_bp(Pb, kb * 4 + t, lane);
            acc[0][t] = wmma_bf16(a0, b, acc[0][t]);
            acc[1][t] = wmma_bf16(a1, b, acc[1][t]);
        }
    }
    int c = lane & 15, half = lane >> 4;
#pragma unroll
    for (int mt = 0; mt < 2; ++mt) {
        int mb = m0 + mt * 16;
        if (mb >= n_rows) break;           // wave-uniform
        bool full = (mb + 16 <= n_rows);
#pragma unroll
        for (int t = 0; t < 4; ++t) {
            int n = t * 16 + c;
            float bv = bias[n];
#pragma unroll
            for (int v = 0; v < 8; ++v) {
                int m = mb + v + 8 * half;
                if (full || m < n_rows) {
                    float val = acc[mt][t][v] + bv;
                    Hout[(long)m * H + n] = val > 0.0f ? val : 0.0f;
                }
            }
        }
    }
}

// h = relu(hx @ P1eff + h0a @ P2eff) where P1eff = (1-b)I + b*w1 (pre-packed),
// hx = (1-ALPHA)*agg, h0a = ALPHA*h0.  Skip connection is folded into the weights.
__global__ void __launch_bounds__(256)
k_gcn2_layer(const float* __restrict__ agg, const float* __restrict__ h0,
             const __bf16* __restrict__ P1, const __bf16* __restrict__ P2,
             float* __restrict__ Hout, int n_rows) {
    int wave = blockIdx.x * (blockDim.x >> 5) + (threadIdx.x >> 5);
    int lane = threadIdx.x & 31;
    int m0 = wave * 32;
    if (m0 >= n_rows) return;
    v8f acc[2][4] = {};
#pragma unroll
    for (int kb = 0; kb < 2; ++kb) {
        v16bf ax0 = load_a(agg, H, m0,      kb * 32, n_rows, lane, 1.0f - ALPHA);
        v16bf ax1 = load_a(agg, H, m0 + 16, kb * 32, n_rows, lane, 1.0f - ALPHA);
        v16bf ah0 = load_a(h0,  H, m0,      kb * 32, n_rows, lane, ALPHA);
        v16bf ah1 = load_a(h0,  H, m0 + 16, kb * 32, n_rows, lane, ALPHA);
#pragma unroll
        for (int t = 0; t < 4; ++t) {
            v16bf b1 = load_bp(P1, kb * 4 + t, lane);
            acc[0][t] = wmma_bf16(ax0, b1, acc[0][t]);
            acc[1][t] = wmma_bf16(ax1, b1, acc[1][t]);
            v16bf b2 = load_bp(P2, kb * 4 + t, lane);
            acc[0][t] = wmma_bf16(ah0, b2, acc[0][t]);
            acc[1][t] = wmma_bf16(ah1, b2, acc[1][t]);
        }
    }
    int c = lane & 15, half = lane >> 4;
#pragma unroll
    for (int mt = 0; mt < 2; ++mt) {
        int mb = m0 + mt * 16;
        if (mb >= n_rows) break;           // wave-uniform
        bool full = (mb + 16 <= n_rows);
#pragma unroll
        for (int t = 0; t < 4; ++t) {
            int n = t * 16 + c;
#pragma unroll
            for (int v = 0; v < 8; ++v) {
                int m = mb + v + 8 * half;
                if (full || m < n_rows) {
                    float val = acc[mt][t][v];
                    Hout[(long)m * H + n] = val > 0.0f ? val : 0.0f;
                }
            }
        }
    }
}

// z[i] = dot(h[i,:], w) + b
__global__ void k_head(const float* __restrict__ h, const float* __restrict__ w,
                       const float* __restrict__ b, float* __restrict__ z, int n) {
    int i = blockIdx.x * blockDim.x + threadIdx.x;
    if (i >= n) return;
    const float* row = h + (long)i * H;
    float s = 0.0f;
#pragma unroll
    for (int k = 0; k < H; k += 4) {
        float4 v = ld4(row + k);
        float4 wv = ld4(w + k);
        s += v.x * wv.x + v.y * wv.y + v.z * wv.z + v.w * wv.w;
    }
    z[i] = s + b[0];
}

// ---------------- launch ----------------

extern "C" void kernel_launch(void* const* d_in, const int* in_sizes, int n_in,
                              void* d_out, int out_size, void* d_ws, size_t ws_size,
                              hipStream_t stream) {
    (void)n_in; (void)out_size; (void)ws_size;
    const float* x   = (const float*)d_in[0];
    const float* xs  = (const float*)d_in[1];
    const int*   ei  = (const int*)d_in[2];
    const float* W0  = (const float*)d_in[3];
    const float* b0  = (const float*)d_in[4];
    const float* W11 = (const float*)d_in[5];
    const float* b11 = (const float*)d_in[6];
    const float* Ws1 = (const float*)d_in[7];
    const float* bs1 = (const float*)d_in[8];
    const float* W3  = (const float*)d_in[9];
    const float* b3  = (const float*)d_in[10];
    const float* cw1  = (const float*)d_in[11];
    const float* cw2  = (const float*)d_in[12];
    const float* c1w1 = (const float*)d_in[13];
    const float* c1w2 = (const float*)d_in[14];

    const int  FIN = 512, FSTR = 58;
    const int  N = in_sizes[0] / FIN;
    const long E = (long)in_sizes[2] / 2;
    const int* srcI = ei;
    const int* dstI = ei + E;

    // workspace layout (256B-aligned chunks)
    auto a64 = [](size_t v) { return (v + 63) & ~(size_t)63; };
    float* wsf = (float*)d_ws;
    size_t p = 0;
    float* dinv = wsf + p; p += a64((size_t)N);
    float* nrm  = wsf + p; p += a64((size_t)E);
    float* h0   = wsf + p; p += a64((size_t)N * H);
    float* hbuf = wsf + p; p += a64((size_t)N * H);
    float* agg  = wsf + p; p += a64((size_t)N * H);
    float* xpad = wsf + p; p += a64((size_t)N * H);   // x_str zero-padded to K=64
    float* Wpad = wsf + p; p += a64((size_t)H * H);   // W_lin11 zero-padded to K=64
    // packed bf16 weights (fragment order)
    __bf16* pk = (__bf16*)(wsf + p);
    size_t q = 0;
    __bf16* pW0  = pk + q; q += a64((size_t)FIN * H);      // 64 frags
    __bf16* pW11 = pk + q; q += a64((size_t)H * H);        // 8 frags
    __bf16* pcw1  = pk + q; q += a64((size_t)4 * H * H);   // effective (1-b)I + b*w
    __bf16* pcw2  = pk + q; q += a64((size_t)4 * H * H);
    __bf16* pc1w1 = pk + q; q += a64((size_t)4 * H * H);
    __bf16* pc1w2 = pk + q; q += a64((size_t)4 * H * H);

    const int TPB = 256;
    const int waves = (N + 31) / 32;                   // 32 rows per wave
    const int gBlk  = (waves + 7) / 8;                 // 8 waves per block
    const int eBlk  = (int)((E + TPB - 1) / TPB);
    const int sBlk  = (int)((E * 16 + TPB - 1) / TPB);
    const int nBlk  = (N + TPB - 1) / TPB;
    const int wBlk  = (H * H + TPB - 1) / TPB;

    // --- pad str-branch operands so all GEMMs have K % 32 == 0 ---
    {
        long tot = (long)N * H;
        k_pad_rows<<<(int)((tot + TPB - 1) / TPB), TPB, 0, stream>>>(xs, xpad, N, FSTR, H);
        k_pad_rows<<<wBlk, TPB, 0, stream>>>(W11, Wpad, H, FSTR, H);
    }

    // --- pre-pack all weights into WMMA B-fragment bf16 layout ---
    k_pack_b<<<(FIN * H + TPB - 1) / TPB, TPB, 0, stream>>>(W0, pW0, FIN, 1, 1.0f, 0.0f);
    k_pack_b<<<wBlk, TPB, 0, stream>>>(Wpad, pW11, H, 1, 1.0f, 0.0f);
    for (int l = 0; l < 4; ++l) {
        long o = (long)l * H * H;
        float beta = logf(THETA / (float)(l + 1) + 1.0f);
        // effective weight folds the GCN2 skip connection: (1-beta)*I + beta*w
        k_pack_b<<<wBlk, TPB, 0, stream>>>(cw1  + o, pcw1  + o, H, 0, beta, 1.0f - beta);
        k_pack_b<<<wBlk, TPB, 0, stream>>>(cw2  + o, pcw2  + o, H, 0, beta, 1.0f - beta);
        k_pack_b<<<wBlk, TPB, 0, stream>>>(c1w1 + o, pc1w1 + o, H, 0, beta, 1.0f - beta);
        k_pack_b<<<wBlk, TPB, 0, stream>>>(c1w2 + o, pc1w2 + o, H, 0, beta, 1.0f - beta);
    }

    // --- normalization ---
    hipMemsetAsync(dinv, 0, (size_t)N * sizeof(float), stream);
    k_degree<<<eBlk, TPB, 0, stream>>>(dstI, dinv, E);
    k_dinv<<<nBlk, TPB, 0, stream>>>(dinv, N);
    k_norm<<<eBlk, TPB, 0, stream>>>(srcI, dstI, dinv, nrm, E);

    // --- branch runner ---
    auto run_branch = [&](const float* xin, int K, const __bf16* pWl, const float* bl,
                          const __bf16* p1, const __bf16* p2,
                          const float* wh, const float* bh, float* zout) {
        k_gemm_relu<<<gBlk, TPB, 0, stream>>>(xin, pWl, bl, h0, N, K);
        const float* hin = h0;
        for (int l = 0; l < 4; ++l) {
            hipMemsetAsync(agg, 0, (size_t)N * H * sizeof(float), stream);
            k_scatter<<<sBlk, TPB, 0, stream>>>(srcI, dstI, nrm, hin, agg, E);
            k_gcn2_layer<<<gBlk, TPB, 0, stream>>>(agg, h0, p1 + (long)l * H * H,
                                                   p2 + (long)l * H * H, hbuf, N);
            hin = hbuf;
        }
        k_head<<<nBlk, TPB, 0, stream>>>(hbuf, wh, bh, zout, N);
    };

    float* out = (float*)d_out;
    run_branch(x,    FIN, pW0,  b0,  pcw1,  pcw2,  Ws1, bs1, out);       // z
    run_branch(xpad, H,   pW11, b11, pc1w1, pc1w2, W3,  b3,  out + N);   // z1
}
